// WorldModelPredictor_20736102105162
// MI455X (gfx1250) — matrix operand order repaired and checked by reading, not compiled, not run on hardware
//
#include <hip/hip_runtime.h>
#include <hip/hip_bf16.h>

// ---------------------------------------------------------------------------
// Types for CDNA5 WMMA (wave32): A/B = 16 x bf16 (8 VGPRs), C/D = 8 x f32.
// ---------------------------------------------------------------------------
typedef __bf16 bf16_t;
typedef __attribute__((ext_vector_type(16))) __bf16 v16bf;
typedef __attribute__((ext_vector_type(8)))  __bf16 v8bf;
typedef __attribute__((ext_vector_type(8)))  float  v8f;

union V16U { v16bf v; v8bf h[2]; };

__device__ __forceinline__ v16bf make_v16(v8bf lo, v8bf hi) {
  V16U u; u.h[0] = lo; u.h[1] = hi; return u.v;
}

// Opaque zero (in an SGPR). Added to weight pointers once per layer call so
// the weight-fragment loads are NOT loop-invariant across the 65 timesteps
// (prevents LICM pre-loading ~960 VGPRs of fragments -> scratch spills),
// while keeping the pointer's global-address-space provenance so the backend
// emits global_load_b128 (LOADcnt only) instead of flat_load (LOAD+DScnt).
__device__ __forceinline__ int opaque_zero() {
  int v = 0;
  asm volatile("" : "+s"(v));
  return v;
}

// A-fragment (16x32, M x K) from row-major [16, stride] bf16 buffer (LDS).
// Layout: lanes 0-15 hold row M=lane, K = {k0..k0+15};
//         lanes 16-31 hold row M=lane-16, K = {k0+16..k0+31}.
__device__ __forceinline__ v16bf load_a_frag(const bf16_t* base, int stride,
                                             int k0, int lane) {
  const int m  = lane & 15;
  const int ks = (lane >> 4) << 4;          // 0 or 16
  const bf16_t* p = base + m * stride + k0 + ks;
  v8bf lo = *(const v8bf*)p;                // 16B ds load
  v8bf hi = *(const v8bf*)(p + 8);          // 16B ds load
  return make_v16(lo, hi);
}

// B-fragment (32x16, K x N) where B = W.T and W is row-major [512, strideK].
// Lane (n = lane&15) needs W[rowbase+n][k0 .. k0+15(+16)] -> 32 contiguous B.
__device__ __forceinline__ v16bf load_b_frag(const bf16_t* __restrict__ W,
                                             int strideK, int rowbase,
                                             int k0, int lane) {
  const int n  = lane & 15;
  const int ko = (lane >> 4) << 4;          // 0 or 16
  const bf16_t* p = W + (size_t)(rowbase + n) * strideK + k0 + ko;
  return *(const v16bf*)p;                  // 32B contiguous global load
}

__device__ __forceinline__ v8f wmma_bf16(v16bf a, v16bf b, v8f c) {
  return __builtin_amdgcn_wmma_f32_16x16x32_bf16(
      /*neg_a=*/false, a, /*neg_b=*/false, b,
      /*c_mod=*/(short)0, c, /*reuse_a=*/false, /*reuse_b=*/false);
}

__device__ __forceinline__ float fast_sigmoid(float x) {
  return 1.0f / (1.0f + __expf(-x));
}
__device__ __forceinline__ float fast_tanh(float x) {
  const float e = __expf(2.0f * x);
  return (e - 1.0f) / (e + 1.0f);
}

// ---------------------------------------------------------------------------
// One LSTM layer for a 32-row batch tile.
//  a_in   : [32, KIN]  bf16 (LDS), layer input
//  h_prev : [32, 128]  bf16 (LDS), previous hidden (also written as h_out)
//  Win    : [512, KIN] bf16 (global), Wih (row-major == B-matrix layout)
//  Whh    : [512, 128] bf16 (global)
//  bias   : [512] f32  (bih + bhh)
//  cst    : per-wave cell state, 2 M-tiles of v8f (registers, persistent)
// Wave w owns hidden channels [16w, 16w+16): gates i/f/g/o for those cols.
// ---------------------------------------------------------------------------
template <int KIN>
__device__ __forceinline__ void lstm_layer(const bf16_t* a_in,
                                           const bf16_t* h_prev,
                                           bf16_t*       h_out,
                                           const bf16_t* Win_in,
                                           const bf16_t* Whh_in,
                                           const float*  bias_in,
                                           v8f* cst, int wave, int lane) {
  // Per-timestep opaque offsets defeat cross-iteration hoisting of the
  // weight loads while preserving global address-space provenance.
  const bf16_t* Win  = Win_in  + opaque_zero();
  const bf16_t* Whh  = Whh_in  + opaque_zero();
  const float*  bias = bias_in + opaque_zero();

  const int ncol = wave * 16 + (lane & 15);

  v8f acc[4][2];
#pragma unroll
  for (int g = 0; g < 4; ++g) {
    const float bv = bias[g * 128 + ncol];
#pragma unroll
    for (int m = 0; m < 2; ++m)
#pragma unroll
      for (int i = 0; i < 8; ++i) acc[g][m][i] = bv;
  }

  // Input contribution: u @ Wih.T
#pragma unroll
  for (int k0 = 0; k0 < KIN; k0 += 32) {
    v16bf a0 = load_a_frag(a_in,            KIN, k0, lane);
    v16bf a1 = load_a_frag(a_in + 16 * KIN, KIN, k0, lane);
#pragma unroll
    for (int g = 0; g < 4; ++g) {
      v16bf bw = load_b_frag(Win, KIN, g * 128 + wave * 16, k0, lane);
      acc[g][0] = wmma_bf16(a0, bw, acc[g][0]);
      acc[g][1] = wmma_bf16(a1, bw, acc[g][1]);
    }
  }
  // Recurrent contribution: h @ Whh.T
#pragma unroll
  for (int k0 = 0; k0 < 128; k0 += 32) {
    v16bf a0 = load_a_frag(h_prev,            128, k0, lane);
    v16bf a1 = load_a_frag(h_prev + 16 * 128, 128, k0, lane);
#pragma unroll
    for (int g = 0; g < 4; ++g) {
      v16bf bw = load_b_frag(Whh, 128, g * 128 + wave * 16, k0, lane);
      acc[g][0] = wmma_bf16(a0, bw, acc[g][0]);
      acc[g][1] = wmma_bf16(a1, bw, acc[g][1]);
    }
  }

  __syncthreads();  // all reads of h_prev complete before overwrite

  // Elementwise gates + cell update; write new h (bf16) back to LDS.
  const int rbase = (lane >> 4) << 3;  // C/D layout: lanes 16-31 hold M=r+8
#pragma unroll
  for (int m = 0; m < 2; ++m) {
#pragma unroll
    for (int r = 0; r < 8; ++r) {
      const float iv = fast_sigmoid(acc[0][m][r]);
      const float fv = fast_sigmoid(acc[1][m][r]);
      const float gv = fast_tanh(acc[2][m][r]);
      const float ov = fast_sigmoid(acc[3][m][r]);
      const float c  = fv * cst[m][r] + iv * gv;
      cst[m][r] = c;
      const float h = ov * fast_tanh(c);
      h_out[(m * 16 + r + rbase) * 128 + ncol] = (bf16_t)h;
    }
  }
  __syncthreads();  // new h visible to all waves
}

// ---------------------------------------------------------------------------
// Prologue: convert/pad weights to bf16, pre-sum biases. One-shot, tiny.
// ---------------------------------------------------------------------------
__global__ void wm_convert_kernel(const float* __restrict__ Wih0,
                                  const float* __restrict__ Whh0,
                                  const float* __restrict__ bih0,
                                  const float* __restrict__ bhh0,
                                  const float* __restrict__ Wih1,
                                  const float* __restrict__ Whh1,
                                  const float* __restrict__ bih1,
                                  const float* __restrict__ bhh1,
                                  const float* __restrict__ Wd,
                                  const float* __restrict__ bd,
                                  bf16_t* __restrict__ w0,   // [512, 96]
                                  bf16_t* __restrict__ wh0,  // [512, 128]
                                  bf16_t* __restrict__ w1,   // [512, 128]
                                  bf16_t* __restrict__ wh1,  // [512, 128]
                                  bf16_t* __restrict__ wd,   // [80, 128]
                                  float* __restrict__ b0,    // [512]
                                  float* __restrict__ b1,    // [512]
                                  float* __restrict__ bdp) { // [80]
  const int i = blockIdx.x * blockDim.x + threadIdx.x;
  if (i < 512 * 96) {  // Wih0: pad K 86 -> 96
    const int r = i / 96, c = i - 96 * r;
    w0[i] = (bf16_t)((c < 86) ? Wih0[r * 86 + c] : 0.0f);
  }
  if (i < 512 * 128) {
    wh0[i] = (bf16_t)Whh0[i];
    w1[i]  = (bf16_t)Wih1[i];
    wh1[i] = (bf16_t)Whh1[i];
  }
  if (i < 80 * 128) {  // Wd: pad OUT 66 -> 80 rows
    const int r = i / 128, c = i - 128 * r;
    wd[i] = (bf16_t)((r < 66) ? Wd[r * 128 + c] : 0.0f);
  }
  if (i < 512) { b0[i] = bih0[i] + bhh0[i]; b1[i] = bih1[i] + bhh1[i]; }
  if (i < 80)  { bdp[i] = (i < 66) ? bd[i] : 0.0f; }
}

// ---------------------------------------------------------------------------
// Main persistent kernel: 256 blocks x 32 batch rows; 65 sequential steps.
// Steps 0..49: scan over x[:,t,:]. Step 50: replay x[:,49,:], emit pose0.
// Steps 51..64: input = concat(pose, cond); cond persists in u_lds[66:86].
// ---------------------------------------------------------------------------
__global__ __launch_bounds__(256) void wm_lstm_kernel(
    const float*  __restrict__ x,
    const bf16_t* __restrict__ Wih0b, const bf16_t* __restrict__ Whh0b,
    const bf16_t* __restrict__ Wih1b, const bf16_t* __restrict__ Whh1b,
    const bf16_t* __restrict__ Wdb,
    const float*  __restrict__ b0, const float* __restrict__ b1,
    const float*  __restrict__ bdp,
    float* __restrict__ out) {
  __shared__ __align__(32) bf16_t u_lds[32 * 96];
  __shared__ __align__(32) bf16_t h1_lds[32 * 128];
  __shared__ __align__(32) bf16_t h2_lds[32 * 128];

  const int tid  = threadIdx.x;
  const int lane = tid & 31;
  const int wave = tid >> 5;
  const size_t row0 = (size_t)blockIdx.x * 32;

  for (int i = tid; i < 32 * 128; i += 256) {
    h1_lds[i] = (bf16_t)0.0f;
    h2_lds[i] = (bf16_t)0.0f;
  }
  v8f c1[2], c2[2];
#pragma unroll
  for (int m = 0; m < 2; ++m)
#pragma unroll
    for (int i = 0; i < 8; ++i) { c1[m][i] = 0.0f; c2[m][i] = 0.0f; }
  __syncthreads();

  for (int step = 0; step < 65; ++step) {
    // ---- stage layer-0 input (scan + replay phases only) ----
    if (step < 51) {
      const int t = (step < 50) ? step : 49;
      for (int i = tid; i < 32 * 96; i += 256) {
        const int r = i / 96, c = i - 96 * r;
        float v = 0.0f;
        if (c < 86) {
          const float* px = &x[((row0 + r) * 50 + t) * 86 + c];
          v = *px;
          if (t < 49) __builtin_prefetch(px + 86, 0, 1);  // next timestep
        }
        u_lds[i] = (bf16_t)v;
      }
    }
    __syncthreads();

    // ---- two LSTM layers (WMMA bf16, f32 accumulate) ----
    lstm_layer<96>(u_lds, h1_lds, h1_lds, Wih0b, Whh0b, b0, c1, wave, lane);
    lstm_layer<128>(h1_lds, h2_lds, h2_lds, Wih1b, Whh1b, b1, c2, wave, lane);

    // ---- dense head + autoregressive feedback ----
    if (step >= 50) {
      if (wave < 5) {  // 5 N-tiles cover padded OUT=80 (66 valid)
        const bf16_t* Wdp = Wdb + opaque_zero();
        const float*  bdl = bdp + opaque_zero();
        const int ncol = wave * 16 + (lane & 15);
        v8f pacc[2];
        const float bv = bdl[ncol];
#pragma unroll
        for (int m = 0; m < 2; ++m)
#pragma unroll
          for (int i = 0; i < 8; ++i) pacc[m][i] = bv;
#pragma unroll
        for (int k0 = 0; k0 < 128; k0 += 32) {
          v16bf a0 = load_a_frag(h2_lds,            128, k0, lane);
          v16bf a1 = load_a_frag(h2_lds + 16 * 128, 128, k0, lane);
          v16bf bw = load_b_frag(Wdp, 128, wave * 16, k0, lane);
          pacc[0] = wmma_bf16(a0, bw, pacc[0]);
          pacc[1] = wmma_bf16(a1, bw, pacc[1]);
        }
        const int sidx  = step - 50;
        const int rbase = (lane >> 4) << 3;
#pragma unroll
        for (int m = 0; m < 2; ++m) {
#pragma unroll
          for (int r = 0; r < 8; ++r) {
            const int row = m * 16 + r + rbase;
            if (ncol < 66) {
              const float v = pacc[m][r];
              out[((row0 + row) * 15 + sidx) * 66 + ncol] = v;  // fp32 result
              u_lds[row * 96 + ncol] = (bf16_t)v;  // feedback (cond persists)
            }
          }
        }
      }
      __syncthreads();  // pose feedback visible before next step's layer 0
    }
  }
}

// ---------------------------------------------------------------------------
// Host entry. Inputs (setup_inputs order):
// 0:x 1:Wih0 2:Whh0 3:bih0 4:bhh0 5:Wih1 6:Whh1 7:bih1 8:bhh1 9:Wd 10:bd
// ---------------------------------------------------------------------------
extern "C" void kernel_launch(void* const* d_in, const int* in_sizes, int n_in,
                              void* d_out, int out_size, void* d_ws,
                              size_t ws_size, hipStream_t stream) {
  const float* x    = (const float*)d_in[0];
  const float* Wih0 = (const float*)d_in[1];
  const float* Whh0 = (const float*)d_in[2];
  const float* bih0 = (const float*)d_in[3];
  const float* bhh0 = (const float*)d_in[4];
  const float* Wih1 = (const float*)d_in[5];
  const float* Whh1 = (const float*)d_in[6];
  const float* bih1 = (const float*)d_in[7];
  const float* bhh1 = (const float*)d_in[8];
  const float* Wd   = (const float*)d_in[9];
  const float* bd   = (const float*)d_in[10];

  char* p = (char*)d_ws;
  bf16_t* w0  = (bf16_t*)p; p += 512 * 96  * sizeof(bf16_t);
  bf16_t* wh0 = (bf16_t*)p; p += 512 * 128 * sizeof(bf16_t);
  bf16_t* w1  = (bf16_t*)p; p += 512 * 128 * sizeof(bf16_t);
  bf16_t* wh1 = (bf16_t*)p; p += 512 * 128 * sizeof(bf16_t);
  bf16_t* wd  = (bf16_t*)p; p += 80  * 128 * sizeof(bf16_t);
  float*  b0  = (float*)p;  p += 512 * sizeof(float);
  float*  b1  = (float*)p;  p += 512 * sizeof(float);
  float*  bdp = (float*)p;  p += 80  * sizeof(float);

  wm_convert_kernel<<<(512 * 128 + 255) / 256, 256, 0, stream>>>(
      Wih0, Whh0, bih0, bhh0, Wih1, Whh1, bih1, bhh1, Wd, bd,
      w0, wh0, w1, wh1, wd, b0, b1, bdp);

  wm_lstm_kernel<<<8192 / 32, 256, 0, stream>>>(
      x, w0, wh0, w1, wh1, wd, b0, b1, bdp, (float*)d_out);
}
